// MultiStepANN_19069654794946
// MI455X (gfx1250) — matrix operand order repaired and checked by reading, not compiled, not run on hardware
//
#include <hip/hip_runtime.h>
#include <hip/hip_bf16.h>

typedef __attribute__((ext_vector_type(16))) _Float16 v16h;
typedef __attribute__((ext_vector_type(8)))  _Float16 v8h;
typedef __attribute__((ext_vector_type(8)))  float    v8f;

static constexpr int BATCH   = 32768;
static constexpr int NCARD   = 60;
static constexpr int F_CARD  = 8;
static constexpr int H_CARD  = 32;
static constexpr int E_CARD  = 16;
static constexpr int F_HAND  = 32;
static constexpr int K_HAND  = NCARD * E_CARD + F_HAND;   // 992
static constexpr int H_HAND  = 256;
static constexpr int E_HAND  = 128;
static constexpr int F_STRAT = 16;
static constexpr int K_STRAT = E_HAND + F_STRAT;          // 144
static constexpr int K_STRATP= 160;                       // padded to /32
static constexpr int H_STRAT = 128;
static constexpr int E_STRAT = 64;
static constexpr int H_Q     = 64;

static inline int cdiv(int a, int b) { return (a + b - 1) / b; }

// ---------------------------------------------------------------------------
// Weight transpose + f32 -> f16 convert: out[n*Kpad + k] = in[k*N + n], zero pad.
// ---------------------------------------------------------------------------
__global__ __launch_bounds__(256)
void transpose_to_f16(const float* __restrict__ in, _Float16* __restrict__ out,
                      int K, int N, int Kpad)
{
    int idx = blockIdx.x * blockDim.x + threadIdx.x;
    if (idx >= N * Kpad) return;
    int n = idx / Kpad;
    int k = idx - n * Kpad;
    out[idx] = (k < K) ? (_Float16)in[(size_t)k * N + n] : (_Float16)0.0f;
}

// ---------------------------------------------------------------------------
// Per-card MLP (K=8 -> 32 -> 16), writes f16 embeddings into hand_in[b, c*16..]
// ---------------------------------------------------------------------------
__global__ __launch_bounds__(256)
void card_mlp(const float* __restrict__ cf,
              const float* __restrict__ cW1, const float* __restrict__ cb1,
              const float* __restrict__ cW2, const float* __restrict__ cb2,
              _Float16* __restrict__ hand_in)
{
    int idx = blockIdx.x * blockDim.x + threadIdx.x;
    if (idx >= BATCH * NCARD) return;
    int b = idx / NCARD;
    int c = idx - b * NCARD;

    float xf[F_CARD];
    const float* x = cf + (size_t)idx * F_CARD;
#pragma unroll
    for (int f = 0; f < F_CARD; ++f) xf[f] = x[f];

    float h[H_CARD];
#pragma unroll
    for (int j = 0; j < H_CARD; ++j) {
        float a = cb1[j];
#pragma unroll
        for (int f = 0; f < F_CARD; ++f) a += xf[f] * cW1[f * H_CARD + j];
        h[j] = a > 0.0f ? a : 0.0f;
    }

    _Float16* o = hand_in + (size_t)b * K_HAND + c * E_CARD;
#pragma unroll
    for (int e = 0; e < E_CARD; ++e) {
        float a = cb2[e];
#pragma unroll 8
        for (int j = 0; j < H_CARD; ++j) a += h[j] * cW2[j * E_CARD + e];
        o[e] = (_Float16)a;
    }
}

// Copy hand_features (f32) into hand_in columns 960..991 (f16)
__global__ __launch_bounds__(256)
void pack_hand_tail(const float* __restrict__ hf, _Float16* __restrict__ hand_in)
{
    int idx = blockIdx.x * blockDim.x + threadIdx.x;
    if (idx >= BATCH * F_HAND) return;
    int b = idx >> 5, j = idx & 31;
    hand_in[(size_t)b * K_HAND + NCARD * E_CARD + j] = (_Float16)hf[idx];
}

// Copy strategy_features into strat_in columns 128..143; zero pad 144..159
__global__ __launch_bounds__(256)
void pack_strat_tail(const float* __restrict__ sf, _Float16* __restrict__ strat_in)
{
    int idx = blockIdx.x * blockDim.x + threadIdx.x;
    if (idx >= BATCH * 32) return;
    int b = idx >> 5, j = idx & 31;
    _Float16 v = (j < F_STRAT) ? (_Float16)sf[(size_t)b * F_STRAT + j] : (_Float16)0.0f;
    strat_in[(size_t)b * K_STRATP + E_HAND + j] = v;
}

// ---------------------------------------------------------------------------
// WMMA GEMM: C[M,N] = act(A[M,K](f16) * B[K,N] + bias), B given transposed
// (BT[n*ldbt + k], f16). Each wave computes a 16 x (16*NBLK) slab with
// v_wmma_f32_16x16x32_f16, f32 accumulation, fused bias (+ optional ReLU).
// A-frag: lane L -> row L&15, 16 contiguous K-values (lanes 16-31: K+16).
// B-frag: lane n -> col n, 16 contiguous K-values (lanes 16-31: K+16).
// D:      vgpr j -> row (lane>>4)*8 + j, col lane&15.
// ---------------------------------------------------------------------------
template<int K, int N, bool RELU, typename OutT>
__global__ __launch_bounds__(256)
void gemm16(const _Float16* __restrict__ A, int lda,
            const _Float16* __restrict__ BT, int ldbt,
            const float* __restrict__ bias,
            OutT* __restrict__ C, int ldc, int M)
{
    constexpr int NBLK = (N >= 64) ? 4 : (N / 16);
    constexpr int NG   = N / (16 * NBLK);

    const int lane = threadIdx.x & 31;
    const int wv   = threadIdx.x >> 5;
    const int tile = blockIdx.x * 8 + wv;              // wave-uniform
    const int total = (M / 16) * NG;
    if (tile >= total) return;                          // uniform per wave: EXEC stays all-1s

    const int tm = tile / NG;
    const int tg = tile - tm * NG;
    const int r  = lane & 15;
    const int hi = lane >> 4;

    const _Float16* ap = A + (size_t)(tm * 16 + r) * lda + hi * 16;
    const _Float16* bp[NBLK];
#pragma unroll
    for (int s = 0; s < NBLK; ++s)
        bp[s] = BT + (size_t)((tg * NBLK + s) * 16 + r) * ldbt + hi * 16;

    v8f acc[NBLK];
#pragma unroll
    for (int s = 0; s < NBLK; ++s) acc[s] = (v8f){0,0,0,0,0,0,0,0};

#pragma unroll 2
    for (int k0 = 0; k0 < K; k0 += 32) {
        v8h alo = *(const v8h*)(ap + k0);
        v8h ahi = *(const v8h*)(ap + k0 + 8);
        v16h a = __builtin_shufflevector(alo, ahi,
                                         0,1,2,3,4,5,6,7,8,9,10,11,12,13,14,15);
#pragma unroll
        for (int s = 0; s < NBLK; ++s) {
            v16h b = *(const v16h*)(bp[s] + k0);
            acc[s] = __builtin_amdgcn_wmma_f32_16x16x32_f16(
                false, a, false, b, (short)0, acc[s], false, false);
        }
    }

#pragma unroll
    for (int s = 0; s < NBLK; ++s) {
        const int ncol = (tg * NBLK + s) * 16 + r;
        const float bv = bias[ncol];
        OutT* cp = C + (size_t)(tm * 16 + hi * 8) * ldc + ncol;
#pragma unroll
        for (int j = 0; j < 8; ++j) {
            float v = acc[s][j] + bv;
            if (RELU) v = v > 0.0f ? v : 0.0f;
            cp[(size_t)j * ldc] = (OutT)v;
        }
    }
}

// ---------------------------------------------------------------------------
// q head: q[b,c] = relu(sq[b,:] + emb[b,c,:] @ qW1[64:,:]) @ qW2 + qb2, masked.
// sq already contains strat_rep @ qW1[:64] + qb1. Weight loads are wave-
// uniform -> scalar-cached.
// ---------------------------------------------------------------------------
__global__ __launch_bounds__(256)
void q_head(const _Float16* __restrict__ hand_in, const float* __restrict__ sq,
            const int* __restrict__ playable,
            const float* __restrict__ qW1, const float* __restrict__ qW2,
            const float* __restrict__ qb2, float* __restrict__ out)
{
    int idx = blockIdx.x * blockDim.x + threadIdx.x;
    if (idx >= BATCH * NCARD) return;
    int b = idx / NCARD;
    int c = idx - b * NCARD;

    float ce[E_CARD];
    const _Float16* ep = hand_in + (size_t)b * K_HAND + c * E_CARD;
#pragma unroll
    for (int e = 0; e < E_CARD; ++e) ce[e] = (float)ep[e];

    const float* sqb   = sq + (size_t)b * H_Q;
    const float* qW1b  = qW1 + (size_t)E_STRAT * H_Q;   // bottom 16 rows
    float qacc = qb2[0];
#pragma unroll 4
    for (int n = 0; n < H_Q; ++n) {
        float a = sqb[n];
#pragma unroll
        for (int e = 0; e < E_CARD; ++e) a += ce[e] * qW1b[e * H_Q + n];
        a = a > 0.0f ? a : 0.0f;
        qacc += a * qW2[n];
    }
    out[idx] = (playable[idx] == 1) ? qacc : 0.0f;
}

// ---------------------------------------------------------------------------
extern "C" void kernel_launch(void* const* d_in, const int* in_sizes, int n_in,
                              void* d_out, int out_size, void* d_ws, size_t ws_size,
                              hipStream_t stream)
{
    const float* cards_features = (const float*)d_in[0];
    const int*   playable       = (const int*)d_in[1];
    const float* hand_features  = (const float*)d_in[2];
    const float* strat_features = (const float*)d_in[3];
    const float* cW1 = (const float*)d_in[4];
    const float* cb1 = (const float*)d_in[5];
    const float* cW2 = (const float*)d_in[6];
    const float* cb2 = (const float*)d_in[7];
    const float* hW1 = (const float*)d_in[8];
    const float* hb1 = (const float*)d_in[9];
    const float* hW2 = (const float*)d_in[10];
    const float* hb2 = (const float*)d_in[11];
    const float* sW1 = (const float*)d_in[12];
    const float* sb1 = (const float*)d_in[13];
    const float* sW2 = (const float*)d_in[14];
    const float* sb2 = (const float*)d_in[15];
    const float* qW1 = (const float*)d_in[16];
    const float* qb1 = (const float*)d_in[17];
    const float* qW2 = (const float*)d_in[18];
    const float* qb2 = (const float*)d_in[19];

    // ---- workspace carve-up (256B aligned regions) ----
    char* w = (char*)d_ws;
    auto alloc = [&](size_t bytes) -> char* {
        char* p = w;
        w += (bytes + 255) & ~(size_t)255;
        return p;
    };
    _Float16* hand_in   = (_Float16*)alloc((size_t)BATCH * K_HAND * 2);   // 65.0 MB
    _Float16* y1        = (_Float16*)alloc((size_t)BATCH * H_HAND * 2);   // 16.8 MB
    _Float16* strat_in  = (_Float16*)alloc((size_t)BATCH * K_STRATP * 2); // 10.5 MB
    _Float16* strat_rep = (_Float16*)alloc((size_t)BATCH * E_STRAT * 2);  //  4.2 MB
    // s1 and sq alias the dead y1 buffer (y1 consumed before s1 is written)
    _Float16* s1 = y1;                                                    //  8.4 MB
    float*    sq = (float*)((char*)y1 + (size_t)BATCH * H_STRAT * 2);     //  8.4 MB
    _Float16* hW1T   = (_Float16*)alloc((size_t)H_HAND * K_HAND * 2);
    _Float16* hW2T   = (_Float16*)alloc((size_t)E_HAND * H_HAND * 2);
    _Float16* sW1Tp  = (_Float16*)alloc((size_t)H_STRAT * K_STRATP * 2);
    _Float16* sW2T   = (_Float16*)alloc((size_t)E_STRAT * H_STRAT * 2);
    _Float16* qW1tT  = (_Float16*)alloc((size_t)H_Q * E_STRAT * 2);
    (void)ws_size; (void)n_in; (void)in_sizes; (void)out_size;

    const int T = 256;

    // ---- stage 0: weight transpose/convert to f16 ----
    transpose_to_f16<<<cdiv(H_HAND * K_HAND, T), T, 0, stream>>>(hW1, hW1T, K_HAND, H_HAND, K_HAND);
    transpose_to_f16<<<cdiv(E_HAND * H_HAND, T), T, 0, stream>>>(hW2, hW2T, H_HAND, E_HAND, H_HAND);
    transpose_to_f16<<<cdiv(H_STRAT * K_STRATP, T), T, 0, stream>>>(sW1, sW1Tp, K_STRAT, H_STRAT, K_STRATP);
    transpose_to_f16<<<cdiv(E_STRAT * H_STRAT, T), T, 0, stream>>>(sW2, sW2T, H_STRAT, E_STRAT, H_STRAT);
    transpose_to_f16<<<cdiv(H_Q * E_STRAT, T), T, 0, stream>>>(qW1, qW1tT, E_STRAT, H_Q, E_STRAT);

    // ---- stage 1: per-card MLP -> hand_in[:, 0:960]; tails ----
    card_mlp<<<cdiv(BATCH * NCARD, T), T, 0, stream>>>(cards_features, cW1, cb1, cW2, cb2, hand_in);
    pack_hand_tail<<<cdiv(BATCH * F_HAND, T), T, 0, stream>>>(hand_features, hand_in);
    pack_strat_tail<<<cdiv(BATCH * 32, T), T, 0, stream>>>(strat_features, strat_in);

    // ---- stage 2: hand_ann ----
    // y1 = relu(hand_in @ hW1 + hb1)                    [B,992]x[992,256]
    gemm16<K_HAND, H_HAND, true, _Float16>
        <<<cdiv((BATCH / 16) * (H_HAND / 64) * 8, 8), T, 0, stream>>>(
            hand_in, K_HAND, hW1T, K_HAND, hb1, y1, H_HAND, BATCH);
    // hand_rep = y1 @ hW2 + hb2 -> strat_in[:, 0:128]   [B,256]x[256,128]
    gemm16<H_HAND, E_HAND, false, _Float16>
        <<<cdiv((BATCH / 16) * (E_HAND / 64) * 8, 8), T, 0, stream>>>(
            y1, H_HAND, hW2T, H_HAND, hb2, strat_in, K_STRATP, BATCH);

    // ---- stage 3: strategy_ann (K padded 144->160 with zeros both sides) ----
    gemm16<K_STRATP, H_STRAT, true, _Float16>
        <<<cdiv((BATCH / 16) * (H_STRAT / 64) * 8, 8), T, 0, stream>>>(
            strat_in, K_STRATP, sW1Tp, K_STRATP, sb1, s1, H_STRAT, BATCH);
    gemm16<H_STRAT, E_STRAT, false, _Float16>
        <<<cdiv((BATCH / 16) * (E_STRAT / 64) * 8, 8), T, 0, stream>>>(
            s1, H_STRAT, sW2T, H_STRAT, sb2, strat_rep, E_STRAT, BATCH);

    // ---- stage 4: sq = strat_rep @ qW1[:64] + qb1 (f32 out) ----
    gemm16<E_STRAT, H_Q, false, float>
        <<<cdiv((BATCH / 16) * (H_Q / 64) * 8, 8), T, 0, stream>>>(
            strat_rep, E_STRAT, qW1tT, E_STRAT, qb1, sq, H_Q, BATCH);

    // ---- stage 5: q head + playable mask ----
    q_head<<<cdiv(BATCH * NCARD, T), T, 0, stream>>>(
        hand_in, sq, playable, qW1, qW2, qb2, (float*)d_out);
}
